// Qwen2MoeLMHead_32375463477425
// MI455X (gfx1250) — compile-verified, matrix operand order checked by reading, not run on hardware
//
#include <hip/hip_runtime.h>

#define HIDDEN 2048
#define VOCAB  151936
#define TOKENS 512
#define EPS    1e-6f
#define KCHUNK 256
#define LDSPAD 8   // bf16 elements of pad per LDS row (16B) -> bank-conflict free

typedef __attribute__((ext_vector_type(16))) __bf16 bf16x16;
typedef __attribute__((ext_vector_type(8)))  __bf16 bf16x8;
typedef __attribute__((ext_vector_type(8)))  float  f32x8;
typedef __attribute__((ext_vector_type(4)))  float  f32x4;

// ---------------------------------------------------------------------------
// Kernel 1: RMSNorm in fp32, emit bf16 activations (WMMA A operand source).
// One block per token, 256 threads, 8 elements/thread.
// ---------------------------------------------------------------------------
__global__ __launch_bounds__(256) void rmsnorm_bf16_kernel(
    const float* __restrict__ x, const float* __restrict__ w,
    __bf16* __restrict__ out)
{
    const int t   = blockIdx.x;
    const int tid = threadIdx.x;
    const float* row = x + (size_t)t * HIDDEN;
    f32x4 v0 = *(const f32x4*)(row + tid * 8);
    f32x4 v1 = *(const f32x4*)(row + tid * 8 + 4);
    float ss = 0.f;
    #pragma unroll
    for (int i = 0; i < 4; ++i) ss += v0[i] * v0[i] + v1[i] * v1[i];

    __shared__ float red[256];
    red[tid] = ss;
    __syncthreads();
    #pragma unroll
    for (int s = 128; s > 0; s >>= 1) {
        if (tid < s) red[tid] += red[tid + s];
        __syncthreads();
    }
    const float inv = rsqrtf(red[0] * (1.0f / HIDDEN) + EPS);

    const float* wr = w + tid * 8;
    bf16x8 o;
    #pragma unroll
    for (int i = 0; i < 4; ++i) {
        o[i]     = (__bf16)((v0[i] * inv) * wr[i]);
        o[4 + i] = (__bf16)((v1[i] * inv) * wr[4 + i]);
    }
    *(bf16x8*)(out + (size_t)t * HIDDEN + tid * 8) = o;
}

// ---------------------------------------------------------------------------
// Helpers for the fused argmax
// ---------------------------------------------------------------------------
__device__ __forceinline__ unsigned long long shfl_xor_u64(unsigned long long v, int mask) {
    unsigned lo = (unsigned)__shfl_xor((int)(unsigned)(v & 0xffffffffull), mask, 32);
    unsigned hi = (unsigned)__shfl_xor((int)(unsigned)(v >> 32), mask, 32);
    return ((unsigned long long)hi << 32) | lo;
}

// ---------------------------------------------------------------------------
// Kernel 2: vocab projection via bf16 WMMA + fused argmax.
// Grid.x = VOCAB/32 (4748). Block = 256 threads = 8 waves.
// Workgroup tile: 32 vocab columns x 512 tokens x K=2048.
//   -> each weight element streamed from HBM exactly ONCE (53us floor).
// W slab (32 cols x 256 K fp32) is converted to bf16 once and staged in LDS.
// Wave w computes M-tiles [4w..4w+3] x 2 N-tiles (8 wmma per K-step).
// ---------------------------------------------------------------------------
__global__ __launch_bounds__(256) void lmhead_wmma_argmax_kernel(
    const __bf16* __restrict__ A,          // [TOKENS][HIDDEN] bf16 (normed)
    const float*  __restrict__ W,          // [VOCAB][HIDDEN] fp32
    unsigned long long* __restrict__ keys) // [TOKENS] packed (sortfloat, ~idx)
{
    __shared__ __bf16 Bs[32][KCHUNK + LDSPAD];   // ~16.5 KB

    const int tid   = threadIdx.x;
    const int lane  = tid & 31;
    const int wave  = tid >> 5;
    const int half  = lane >> 4;   // 0: lanes 0-15, 1: lanes 16-31
    const int l15   = lane & 15;
    const int vbase = blockIdx.x * 32;

    f32x8 acc[4][2] = {};

    // cooperative W-load mapping: thread -> (column, 32 consecutive K)
    const int cCol = tid >> 3;        // 0..31
    const int cK0  = (tid & 7) * 32;  // 0..224

    for (int kc = 0; kc < HIDDEN; kc += KCHUNK) {
        __syncthreads();   // protect Bs from previous iteration's readers
        {
            const float* src = W + (size_t)(vbase + cCol) * HIDDEN + kc + cK0;
            #pragma unroll
            for (int j = 0; j < 4; ++j) {
                f32x4 f0 = *(const f32x4*)(src + j * 8);
                f32x4 f1 = *(const f32x4*)(src + j * 8 + 4);
                bf16x8 b;
                #pragma unroll
                for (int i = 0; i < 4; ++i) {
                    b[i]     = (__bf16)f0[i];
                    b[4 + i] = (__bf16)f1[i];
                }
                *(bf16x8*)&Bs[cCol][cK0 + j * 8] = b;
            }
        }
        __syncthreads();

        #pragma unroll 2
        for (int ks = 0; ks < KCHUNK; ks += 32) {
            // A fragments: 16x32 bf16, lane<16 row=l15 K{0..7,16..23}, lane>=16 K{8..15,24..31}
            bf16x16 af[4];
            #pragma unroll
            for (int m = 0; m < 4; ++m) {
                const int row = (wave * 4 + m) * 16 + l15;
                const __bf16* ap = A + (size_t)row * HIDDEN + kc + ks + half * 8;
                bf16x8 a0 = *(const bf16x8*)ap;
                bf16x8 a1 = *(const bf16x8*)(ap + 16);
                #pragma unroll
                for (int i = 0; i < 8; ++i) { af[m][i] = a0[i]; af[m][8 + i] = a1[i]; }
            }
            // B fragments from LDS (column-major per lane: lane = N, elems = K)
            bf16x16 bfr[2];
            #pragma unroll
            for (int n = 0; n < 2; ++n) {
                const __bf16* bp = &Bs[n * 16 + l15][ks + half * 8];
                bf16x8 b0 = *(const bf16x8*)bp;
                bf16x8 b1 = *(const bf16x8*)(bp + 16);
                #pragma unroll
                for (int i = 0; i < 8; ++i) { bfr[n][i] = b0[i]; bfr[n][8 + i] = b1[i]; }
            }
            #pragma unroll
            for (int m = 0; m < 4; ++m)
                #pragma unroll
                for (int n = 0; n < 2; ++n)
                    acc[m][n] = __builtin_amdgcn_wmma_f32_16x16x32_bf16(
                        false, af[m], false, bfr[n], (short)0, acc[m][n], false, false);
        }
    }

    // Fused argmax epilogue.
    // D layout: VGPR r, lanes 0-15 -> (M=r, N=lane); lanes 16-31 -> (M=r+8, N=lane-16)
    #pragma unroll
    for (int m = 0; m < 4; ++m) {
        #pragma unroll
        for (int r = 0; r < 8; ++r) {
            unsigned long long key = 0ull;
            #pragma unroll
            for (int n = 0; n < 2; ++n) {
                float f = acc[m][n][r];
                unsigned u = __float_as_uint(f);
                u = (u & 0x80000000u) ? ~u : (u | 0x80000000u);  // order-preserving map
                int v = vbase + n * 16 + l15;
                unsigned long long k =
                    ((unsigned long long)u << 32) | (unsigned)(~v); // ties -> lowest idx
                if (k > key) key = k;
            }
            #pragma unroll
            for (int off = 8; off > 0; off >>= 1) {
                unsigned long long o = shfl_xor_u64(key, off);
                if (o > key) key = o;
            }
            if (l15 == 0) {
                int token = (wave * 4 + m) * 16 + half * 8 + r;
                atomicMax(&keys[token], key);
            }
        }
    }
}

// ---------------------------------------------------------------------------
// Kernel 0/3: init packed keys, extract argmax index.
// ---------------------------------------------------------------------------
__global__ void init_keys_kernel(unsigned long long* keys) {
    int t = blockIdx.x * blockDim.x + threadIdx.x;
    if (t < TOKENS) keys[t] = 0ull;
}

__global__ void extract_kernel(const unsigned long long* __restrict__ keys,
                               int* __restrict__ out) {
    int t = blockIdx.x * blockDim.x + threadIdx.x;
    if (t < TOKENS) out[t] = (int)~(unsigned)(keys[t] & 0xffffffffull);
}

// ---------------------------------------------------------------------------
extern "C" void kernel_launch(void* const* d_in, const int* in_sizes, int n_in,
                              void* d_out, int out_size, void* d_ws, size_t ws_size,
                              hipStream_t stream) {
    const float* hidden = (const float*)d_in[0];   // [512, 2048] fp32
    const float* normw  = (const float*)d_in[1];   // [2048] fp32
    const float* lmw    = (const float*)d_in[2];   // [151936, 2048] fp32

    // workspace layout: [0,4096) u64 argmax keys; [4096, 4096+2MB) bf16 A
    unsigned long long* keys = (unsigned long long*)d_ws;
    __bf16* A = (__bf16*)((char*)d_ws + 4096);

    init_keys_kernel<<<(TOKENS + 255) / 256, 256, 0, stream>>>(keys);
    rmsnorm_bf16_kernel<<<TOKENS, 256, 0, stream>>>(hidden, normw, A);
    lmhead_wmma_argmax_kernel<<<VOCAB / 32, 256, 0, stream>>>(A, lmw, keys);
    extract_kernel<<<(TOKENS + 255) / 256, 256, 0, stream>>>(keys, (int*)d_out);
}
